// HyperGCN_70918499991750
// MI455X (gfx1250) — compile-verified, matrix operand order checked by reading, not run on hardware
//
#include <hip/hip_runtime.h>

typedef __attribute__((ext_vector_type(16))) _Float16 v16h;
typedef __attribute__((ext_vector_type(8)))  _Float16 v8h;
typedef __attribute__((ext_vector_type(8)))  float    v8f;

#define N_NODES 100000
#define N_EDGES 20000
#define NNZ     1600000

// ---------------- prep kernels ----------------

__global__ __launch_bounds__(256) void cvt_f32_to_f16(const float* __restrict__ src,
                                                      _Float16* __restrict__ dst, int n) {
  int i = blockIdx.x * blockDim.x + threadIdx.x;
  if (i < n) dst[i] = (_Float16)src[i];
}

// W: [128][N] row-major (fi x fo). Wt: [N][128] fp16 (so B-fragment loads are contiguous).
__global__ __launch_bounds__(256) void transpose_w(const float* __restrict__ W,
                                                   _Float16* __restrict__ Wt, int N) {
  int i = blockIdx.x * blockDim.x + threadIdx.x;
  if (i < N * 128) {
    int n = i >> 7, k = i & 127;
    Wt[i] = (_Float16)W[k * N + n];
  }
}

__global__ __launch_bounds__(256) void degrees_kernel(const int* __restrict__ node_idx,
                                                      const int* __restrict__ edge_idx,
                                                      float* __restrict__ deg_n,
                                                      float* __restrict__ deg_e) {
  int j = blockIdx.x * blockDim.x + threadIdx.x;
  if (j < NNZ) {
    atomicAdd(&deg_n[node_idx[j]], 1.0f);
    atomicAdd(&deg_e[edge_idx[j]], 1.0f);
  }
}

__global__ __launch_bounds__(256) void recip_kernel(const float* __restrict__ deg,
                                                    float* __restrict__ inv, int n) {
  int i = blockIdx.x * blockDim.x + threadIdx.x;
  if (i < n) {
    float d = deg[i];
    inv[i] = (d > 0.0f) ? (1.0f / d) : 0.0f;
  }
}

// ---------------- WMMA GEMM: C[M][N] = A16[M][128] * W[128][N]  (Wt = W^T fp16) -------

template <int NT>  // N = NT*16
__global__ __launch_bounds__(256) void gemm_wmma(const _Float16* __restrict__ A,
                                                 const _Float16* __restrict__ Wt,
                                                 float* __restrict__ C, int M) {
  constexpr int N = NT * 16;
  const int lane = threadIdx.x & 31;
  const int wave = threadIdx.x >> 5;
  const int tile = blockIdx.x * 8 + wave;
  if (tile * 16 >= M) return;  // wave-uniform: EXEC stays all-ones for WMMA waves

  const int half_sel = lane >> 4;   // 0: K-offset 0, 1: K-offset 8 (per ISA 16-bit A/B layout)
  const int mrow = lane & 15;

  // A fragments for 4 k-steps of 32
  const _Float16* arow = A + (size_t)(tile * 16 + mrow) * 128 + half_sel * 8;
  v16h afrag[4];
#pragma unroll
  for (int ks = 0; ks < 4; ++ks) {
    v8h lo = *(const v8h*)(arow + ks * 32);
    v8h hi = *(const v8h*)(arow + ks * 32 + 16);
    v16h a;
#pragma unroll
    for (int q = 0; q < 8; ++q) { a[q] = lo[q]; a[q + 8] = hi[q]; }
    afrag[ks] = a;
  }

  v8f acc[NT];
#pragma unroll
  for (int t = 0; t < NT; ++t) acc[t] = (v8f){};

#pragma unroll
  for (int t = 0; t < NT; ++t) {
    const _Float16* brow = Wt + (size_t)(t * 16 + mrow) * 128 + half_sel * 8;
    v8f c = acc[t];
#pragma unroll
    for (int ks = 0; ks < 4; ++ks) {
      v8h lo = *(const v8h*)(brow + ks * 32);
      v8h hi = *(const v8h*)(brow + ks * 32 + 16);
      v16h b;
#pragma unroll
      for (int q = 0; q < 8; ++q) { b[q] = lo[q]; b[q + 8] = hi[q]; }
      c = __builtin_amdgcn_wmma_f32_16x16x32_f16(false, afrag[ks], false, b,
                                                 (short)0, c, false, false);
    }
    acc[t] = c;
  }

  // D layout: lane<16 -> rows 0..7, lane>=16 -> rows 8..15; col = t*16 + lane%16
  const int rbase = tile * 16 + half_sel * 8;
#pragma unroll
  for (int t = 0; t < NT; ++t) {
#pragma unroll
    for (int r = 0; r < 8; ++r) {
      C[(size_t)(rbase + r) * N + t * 16 + mrow] = acc[t][r];
    }
  }
}

// ---------------- scatter phases ----------------
// shift: log2(F/4). Each thread handles 4 consecutive floats of one incidence.

__global__ __launch_bounds__(256) void scatter_n2e(const float* __restrict__ h,
                                                   const int* __restrict__ node_idx,
                                                   const int* __restrict__ edge_idx,
                                                   float* __restrict__ eacc, int shift) {
  long long tid = (long long)blockIdx.x * blockDim.x + threadIdx.x;
  long long j = tid >> shift;
  if (j >= NNZ) return;
  int f = (int)(tid & ((1 << shift) - 1)) << 2;
  int ni = node_idx[j], ei = edge_idx[j];
  float4 v = *(const float4*)(h + ((size_t)ni << (shift + 2)) + f);
  float* dst = eacc + ((size_t)ei << (shift + 2)) + f;
  atomicAdd(dst + 0, v.x);
  atomicAdd(dst + 1, v.y);
  atomicAdd(dst + 2, v.z);
  atomicAdd(dst + 3, v.w);
}

__global__ __launch_bounds__(256) void scatter_e2n(const float* __restrict__ eacc,
                                                   const int* __restrict__ node_idx,
                                                   const int* __restrict__ edge_idx,
                                                   const float* __restrict__ Be,
                                                   float* __restrict__ nacc, int shift) {
  long long tid = (long long)blockIdx.x * blockDim.x + threadIdx.x;
  long long j = tid >> shift;
  if (j >= NNZ) return;
  int f = (int)(tid & ((1 << shift) - 1)) << 2;
  int ni = node_idx[j], ei = edge_idx[j];
  float be = Be[ei];
  float4 v = *(const float4*)(eacc + ((size_t)ei << (shift + 2)) + f);
  float* dst = nacc + ((size_t)ni << (shift + 2)) + f;
  atomicAdd(dst + 0, be * v.x);
  atomicAdd(dst + 1, be * v.y);
  atomicAdd(dst + 2, be * v.z);
  atomicAdd(dst + 3, be * v.w);
}

// out = relu?( D[n]*acc + bias ), written as fp16 (next layer input) and/or fp32 (final)
__global__ __launch_bounds__(256) void finalize_kernel(const float* __restrict__ acc,
                                                       const float* __restrict__ Dn,
                                                       const float* __restrict__ bias,
                                                       int fshift, int do_relu,
                                                       _Float16* __restrict__ h16,
                                                       float* __restrict__ outf) {
  long long tid = (long long)blockIdx.x * blockDim.x + threadIdx.x;
  long long total = (long long)N_NODES << fshift;
  if (tid >= total) return;
  int n = (int)(tid >> fshift);
  int f = (int)(tid & ((1 << fshift) - 1));
  float v = Dn[n] * acc[tid] + bias[f];
  if (do_relu) v = fmaxf(v, 0.0f);
  if (h16) h16[tid] = (_Float16)v;
  if (outf) outf[tid] = v;
}

// ---------------- host launch ----------------

extern "C" void kernel_launch(void* const* d_in, const int* in_sizes, int n_in,
                              void* d_out, int out_size, void* d_ws, size_t ws_size,
                              hipStream_t stream) {
  const float* x  = (const float*)d_in[0];
  const float* W1 = (const float*)d_in[1];
  const float* b1 = (const float*)d_in[2];
  const float* W2 = (const float*)d_in[3];
  const float* b2 = (const float*)d_in[4];
  const float* W3 = (const float*)d_in[5];
  const float* b3 = (const float*)d_in[6];
  const int* node_idx = (const int*)d_in[7];
  const int* edge_idx = (const int*)d_in[8];
  float* out = (float*)d_out;

  // workspace layout (256B aligned). big is shared: GEMM output (h), then node accumulator.
  char* p = (char*)d_ws;
  auto alloc = [&](size_t bytes) -> char* {
    char* r = p;
    p += (bytes + 255) & ~(size_t)255;
    return r;
  };
  _Float16* x16  = (_Float16*)alloc((size_t)N_NODES * 128 * 2);  // layer input fp16 (reused)
  _Float16* wt1  = (_Float16*)alloc(128 * 128 * 2);
  _Float16* wt2  = (_Float16*)alloc(128 * 128 * 2);
  _Float16* wt3  = (_Float16*)alloc(64 * 128 * 2);
  float* big     = (float*)alloc((size_t)N_NODES * 128 * 4);     // h / node accumulator
  float* eacc    = (float*)alloc((size_t)N_EDGES * 128 * 4);
  float* deg_n   = (float*)alloc((size_t)N_NODES * 4);
  float* deg_e   = (float*)alloc((size_t)N_EDGES * 4);
  float* Dn      = (float*)alloc((size_t)N_NODES * 4);
  float* Be      = (float*)alloc((size_t)N_EDGES * 4);
  (void)ws_size; (void)in_sizes; (void)n_in; (void)out_size;

  // ---- prep ----
  {
    int n = N_NODES * 128;
    cvt_f32_to_f16<<<(n + 255) / 256, 256, 0, stream>>>(x, x16, n);
  }
  transpose_w<<<(128 * 128 + 255) / 256, 256, 0, stream>>>(W1, wt1, 128);
  transpose_w<<<(128 * 128 + 255) / 256, 256, 0, stream>>>(W2, wt2, 128);
  transpose_w<<<(64 * 128 + 255) / 256, 256, 0, stream>>>(W3, wt3, 64);

  hipMemsetAsync(deg_n, 0, (size_t)N_NODES * 4, stream);
  hipMemsetAsync(deg_e, 0, (size_t)N_EDGES * 4, stream);
  degrees_kernel<<<(NNZ + 255) / 256, 256, 0, stream>>>(node_idx, edge_idx, deg_n, deg_e);
  recip_kernel<<<(N_NODES + 255) / 256, 256, 0, stream>>>(deg_n, Dn, N_NODES);
  recip_kernel<<<(N_EDGES + 255) / 256, 256, 0, stream>>>(deg_e, Be, N_EDGES);

  // ---- one hypergraph conv layer ----
  auto layer = [&](const _Float16* in16, const _Float16* wt, const float* bias, int F,
                   bool relu, _Float16* h16_out, float* f32_out) {
    const int tiles = N_NODES / 16;  // 6250 exactly
    if (F == 128)
      gemm_wmma<8><<<(tiles + 7) / 8, 256, 0, stream>>>(in16, wt, big, N_NODES);
    else
      gemm_wmma<4><<<(tiles + 7) / 8, 256, 0, stream>>>(in16, wt, big, N_NODES);

    const int shift = (F == 128) ? 5 : 4;  // log2(F/4)
    hipMemsetAsync(eacc, 0, (size_t)N_EDGES * F * 4, stream);
    long long t1 = (long long)NNZ << shift;
    scatter_n2e<<<(unsigned)((t1 + 255) / 256), 256, 0, stream>>>(big, node_idx, edge_idx,
                                                                  eacc, shift);
    hipMemsetAsync(big, 0, (size_t)N_NODES * F * 4, stream);
    scatter_e2n<<<(unsigned)((t1 + 255) / 256), 256, 0, stream>>>(eacc, node_idx, edge_idx,
                                                                  Be, big, shift);
    const int fshift = (F == 128) ? 7 : 6;
    long long t2 = (long long)N_NODES << fshift;
    finalize_kernel<<<(unsigned)((t2 + 255) / 256), 256, 0, stream>>>(
        big, Dn, bias, fshift, relu ? 1 : 0, h16_out, f32_out);
  };

  layer(x16, wt1, b1, 128, true, x16, nullptr);   // h1 -> x16 (in-place across stream order)
  layer(x16, wt2, b2, 128, true, x16, nullptr);   // h2 -> x16
  layer(x16, wt3, b3, 64, false, nullptr, out);   // final fp32 output
}
